// HourglassTokenizer_66365834657795
// MI455X (gfx1250) — matrix operand order repaired and checked by reading, not compiled, not run on hardware
//
#include <hip/hip_runtime.h>

#define DEV __device__ __forceinline__

typedef __attribute__((ext_vector_type(16))) __bf16 v16bf;
typedef __attribute__((ext_vector_type(8)))  float  v8f;

union BF16Frag { uint4 u4[2]; v16bf v; };

static constexpr int B_ = 64, T_ = 243, J_ = 17, C_ = 256, TR_ = 60, K_ = 30;
static constexpr size_t MROWS = (size_t)B_ * T_ * J_;     // 264384

// ---------------- workspace layout (bytes, all 256-aligned) ----------------
static constexpr size_t H_OFF    = 0;
static constexpr size_t H_BYTES  = MROWS * C_ * 2;                 // 135.4 MB (h bf16)
static constexpr size_t XD_OFF   = H_OFF + H_BYTES;
static constexpr size_t XD_BYTES = (size_t)B_ * TR_ * J_ * C_ * 4; // 66.8 MB (fp32, L2-resident)
static constexpr size_t SC_OFF   = XD_OFF + XD_BYTES;
static constexpr size_t SC_BYTES = (size_t)B_ * TR_ * 4;
static constexpr size_t TI_OFF   = SC_OFF + ((SC_BYTES + 255) & ~(size_t)255);
static constexpr size_t TI_BYTES = (size_t)B_ * K_ * 4;
static constexpr size_t W1_OFF   = TI_OFF + ((TI_BYTES + 255) & ~(size_t)255);
static constexpr size_t W2_OFF   = W1_OFF + (size_t)C_ * C_ * 2;

// ---------------- helpers ----------------
DEV unsigned short f2bf(float f) {                 // fp32 -> bf16 RNE
  unsigned u = __float_as_uint(f);
  unsigned r = 0x7FFFu + ((u >> 16) & 1u);
  return (unsigned short)((u + r) >> 16);
}

DEV float bf2f(unsigned short u) {
  return __uint_as_float(((unsigned)u) << 16);
}

DEV float gelu_erf(float v) {                      // torch GELU default (erf)
  return 0.5f * v * (1.0f + erff(v * 0.70710678118654752440f));
}

// XOR-swizzled A-tile addressing: 16B chunk column XOR (row&15)<<4.
// Each 16-lane half of a ds_load_b128 sweeps all 64 LDS banks exactly once.
DEV unsigned short* atile_at(unsigned short* base, int row, int kbyte) {
  int sw = kbyte ^ ((row & 15) << 4);
  return base + row * 256 + (sw >> 1);
}

// ---------------- kernels ----------------

// weight fp32 [K][N] -> bf16 transposed [N][K]
__global__ __launch_bounds__(256) void wconv_kernel(const float* __restrict__ W,
                                                    unsigned short* __restrict__ Wt) {
  int n = blockIdx.x, k = threadIdx.x;
  Wt[n * 256 + k] = f2bf(W[k * 256 + n]);
}

// Fused (prologue -> LayerNorm -> bf16 A-tile in LDS) -> WMMA GEMM -> epilogue.
// MODE 0: A row = x row (feature_processor);          epilogue: GELU -> h (bf16)
// MODE 1: A row = lerp(gather(xd, top_idx)) (interp); epilogue: GELU + residual -> out (fp32)
// Block = 256 threads = 8 waves = 64 rows x 256 cols.
// Wave tile = 32(M) x 64(N): 8 f32 accumulators (2 m-tiles share each B fragment),
// 8 K-steps of 32 -> 64 v_wmma per wave.
template <int MODE>
__global__ __launch_bounds__(256) void gemm_fused_kernel(
    const float* __restrict__ src,        // MODE0: x ; MODE1: xd
    const int* __restrict__ ti,           // MODE1: top indices, else unused
    const float* __restrict__ g,          // LN gamma
    const float* __restrict__ bta,        // LN beta
    const unsigned short* __restrict__ Wt,// bf16 W^T [N][K]
    const float* __restrict__ bias,
    const float* __restrict__ residual,   // MODE1: x
    void* __restrict__ out_p) {           // MODE0: ushort* ; MODE1: float*
  __shared__ unsigned short Atile[64 * 256];   // 32 KB, swizzled

  const int lane = threadIdx.x & 31;
  const int wave = threadIdx.x >> 5;
  const int blockRow = blockIdx.x * 64;

  // ---- phase 1: build LayerNorm'd bf16 A-tile (wave per row, 8 rows/wave) ----
#pragma unroll
  for (int i = 0; i < 8; i++) {
    const int lrow = wave * 8 + i;
    const int grow = blockRow + lrow;
    float v[8];
    if (MODE == 0) {
      const float* p = src + (size_t)grow * 256 + lane * 8;
      float4 a0 = *(const float4*)p;
      float4 a1 = *(const float4*)(p + 4);
      v[0] = a0.x; v[1] = a0.y; v[2] = a0.z; v[3] = a0.w;
      v[4] = a1.x; v[5] = a1.y; v[6] = a1.z; v[7] = a1.w;
    } else {
      const int j = grow % J_;
      const int t = (grow / J_) % T_;
      const int b = grow / (J_ * T_);
      float src_t = fminf(fmaxf(((float)t + 0.5f) * (30.0f / 243.0f) - 0.5f, 0.0f), 29.0f);
      int   i0 = (int)src_t;
      int   i1 = min(i0 + 1, K_ - 1);
      float w  = src_t - (float)i0;
      float iw = 1.0f - w;
      int k0i = ti[b * K_ + i0];
      int k1i = ti[b * K_ + i1];
      const float* p0 = src + (((size_t)(b * TR_ + k0i)) * J_ + j) * 256 + lane * 8;
      const float* p1 = src + (((size_t)(b * TR_ + k1i)) * J_ + j) * 256 + lane * 8;
      float4 a0 = *(const float4*)p0, a1 = *(const float4*)(p0 + 4);
      float4 c0 = *(const float4*)p1, c1 = *(const float4*)(p1 + 4);
      v[0] = a0.x * iw + c0.x * w; v[1] = a0.y * iw + c0.y * w;
      v[2] = a0.z * iw + c0.z * w; v[3] = a0.w * iw + c0.w * w;
      v[4] = a1.x * iw + c1.x * w; v[5] = a1.y * iw + c1.y * w;
      v[6] = a1.z * iw + c1.z * w; v[7] = a1.w * iw + c1.w * w;
    }
    // LayerNorm across the wave (row = 256 channels, lane holds 8 contiguous)
    float s1 = 0.f, s2 = 0.f;
#pragma unroll
    for (int q = 0; q < 8; q++) { s1 += v[q]; s2 += v[q] * v[q]; }
#pragma unroll
    for (int off = 16; off > 0; off >>= 1) {
      s1 += __shfl_xor(s1, off, 32);
      s2 += __shfl_xor(s2, off, 32);
    }
    float m   = s1 * (1.0f / 256.0f);
    float var = s2 * (1.0f / 256.0f) - m * m;
    float rs  = rsqrtf(var + 1e-5f);
    unsigned short o[8];
#pragma unroll
    for (int q = 0; q < 8; q++) {
      int c = lane * 8 + q;
      o[q] = f2bf((v[q] - m) * rs * g[c] + bta[c]);
    }
    *(uint4*)atile_at(Atile, lrow, lane * 16) = *(const uint4*)o;
  }
  __syncthreads();

  // ---- phase 2: WMMA K-loop. A from LDS (2 m-tiles), B (W^T, L2-resident) from
  // global, each B fragment reused by both m-tiles.
  const int mpair = wave & 1;   // rows mpair*32 .. +32 (2 m-tiles of 16)
  const int ngrp  = wave >> 1;  // 4 N-groups of 64 cols
  const int m     = lane & 15;
  const int half  = lane >> 4;
  const int arow0 = mpair * 32 + m;
  const int arow1 = mpair * 32 + 16 + m;

  v8f z = {0.f, 0.f, 0.f, 0.f, 0.f, 0.f, 0.f, 0.f};
  v8f acc[8] = {z, z, z, z, z, z, z, z};   // [mt*4 + nt]

#pragma unroll
  for (int k0 = 0; k0 < 256; k0 += 32) {
    // A fragments (16x32 bf16): lane(m,half): K = half*8+[0..7] and 16+half*8+[0..7]
    BF16Frag a0, a1;
    a0.u4[0] = *(const uint4*)atile_at(Atile, arow0, k0 * 2 + 16 * half);
    a0.u4[1] = *(const uint4*)atile_at(Atile, arow0, k0 * 2 + 32 + 16 * half);
    a1.u4[0] = *(const uint4*)atile_at(Atile, arow1, k0 * 2 + 16 * half);
    a1.u4[1] = *(const uint4*)atile_at(Atile, arow1, k0 * 2 + 32 + 16 * half);
#pragma unroll
    for (int nt = 0; nt < 4; nt++) {
      // B fragment (32x16 bf16): lane(n=lane&15,half): K = half*8+[0..7] and 16+half*8+[0..7] of col n
      int ncol = ngrp * 64 + nt * 16 + (lane & 15);
      const unsigned short* bcol = Wt + (size_t)ncol * 256 + k0 + half * 8;
      BF16Frag bfg;
      bfg.u4[0] = *(const uint4*)(bcol);
      bfg.u4[1] = *(const uint4*)(bcol + 16);
      acc[nt] = __builtin_amdgcn_wmma_f32_16x16x32_bf16(
          false, a0.v, false, bfg.v, (short)0, acc[nt], false, false);
      acc[4 + nt] = __builtin_amdgcn_wmma_f32_16x16x32_bf16(
          false, a1.v, false, bfg.v, (short)0, acc[4 + nt], false, false);
    }
  }

  // ---- epilogue: C/D layout -> row = base + 8*(lane>>4) + r, col = base + (lane&15)
#pragma unroll
  for (int mt = 0; mt < 2; mt++) {
#pragma unroll
    for (int nt = 0; nt < 4; nt++) {
      int col = ngrp * 64 + nt * 16 + (lane & 15);
      float bv = bias[col];
#pragma unroll
      for (int r = 0; r < 8; r++) {
        int row = blockRow + mpair * 32 + mt * 16 + half * 8 + r;
        size_t idx = (size_t)row * 256 + col;
        float y = gelu_erf(acc[mt * 4 + nt][r] + bv);
        if (MODE == 0) {
          ((unsigned short*)out_p)[idx] = f2bf(y);
        } else {
          ((float*)out_p)[idx] = y + residual[idx];
        }
      }
    }
  }
}

// adaptive avg-pool over T: 243 -> 60 variable-size bins (h is bf16)
__global__ __launch_bounds__(256) void pool_kernel(const unsigned short* __restrict__ h,
                                                   float* __restrict__ xd) {
  int id = blockIdx.x;                 // b*60*17 + tr*17 + j
  int j  = id % J_;
  int tr = (id / J_) % TR_;
  int b  = id / (J_ * TR_);
  int start = (tr * T_) / TR_;
  int end   = ((tr + 1) * T_ + TR_ - 1) / TR_;
  int c = threadIdx.x;
  float s = 0.f;
  for (int t = start; t < end; t++)
    s += bf2f(h[(((size_t)b * T_ + t) * J_ + j) * 256 + c]);
  xd[(size_t)id * 256 + c] = s / (float)(end - start);
}

// J-mean + scorer MLP (256->128->1) + sigmoid; one block per (b,tr)
__global__ __launch_bounds__(256) void scorer_kernel(const float* __restrict__ xd,
                                                     const float* __restrict__ w1,
                                                     const float* __restrict__ b1,
                                                     const float* __restrict__ w2,
                                                     const float* __restrict__ b2,
                                                     float* __restrict__ score) {
  __shared__ float pm[256];
  __shared__ float h1[128];
  int id = blockIdx.x;                 // b*60 + tr
  int c  = threadIdx.x;
  const float* base = xd + ((size_t)id * J_) * 256;
  float s = 0.f;
  for (int j = 0; j < J_; j++) s += base[(size_t)j * 256 + c];
  pm[c] = s * (1.0f / (float)J_);
  __syncthreads();
  if (c < 128) {
    float d = b1[c];
    for (int k = 0; k < 256; k++) d += pm[k] * w1[k * 128 + c];
    h1[c] = gelu_erf(d);
  }
  __syncthreads();
  if (c == 0) {
    float d = b2[0];
    for (int k = 0; k < 128; k++) d += h1[k] * w2[k];
    score[id] = 1.0f / (1.0f + expf(-d));
  }
}

// top-K=30 of 60 by rank counting (matches lax.top_k tie-break), indices ascending
__global__ void topk_kernel(const float* __restrict__ score, int* __restrict__ ti) {
  __shared__ float s[TR_];
  __shared__ int   sel[64];
  int b = blockIdx.x, i = threadIdx.x;
  if (i < TR_) s[i] = score[b * TR_ + i];
  __syncthreads();
  int isSel = 0;
  if (i < TR_) {
    float si = s[i];
    int rank = 0;
    for (int j = 0; j < TR_; j++) {
      float sj = s[j];
      rank += (sj > si) || (sj == si && j < i);
    }
    isSel = rank < K_;
  }
  sel[i] = isSel;
  __syncthreads();
  if (i < TR_ && isSel) {
    int pos = 0;
    for (int j = 0; j < i; j++) pos += sel[j];
    ti[b * K_ + pos] = i;
  }
}

// ---------------- launch ----------------
extern "C" void kernel_launch(void* const* d_in, const int* in_sizes, int n_in,
                              void* d_out, int out_size, void* d_ws, size_t ws_size,
                              hipStream_t stream) {
  (void)in_sizes; (void)n_in; (void)out_size; (void)ws_size;
  const float* x       = (const float*)d_in[0];
  const float* fp_g    = (const float*)d_in[1];
  const float* fp_b    = (const float*)d_in[2];
  const float* fp_w    = (const float*)d_in[3];
  const float* fp_bias = (const float*)d_in[4];
  const float* sc_w1   = (const float*)d_in[5];
  const float* sc_b1   = (const float*)d_in[6];
  const float* sc_w2   = (const float*)d_in[7];
  const float* sc_b2   = (const float*)d_in[8];
  const float* op_g    = (const float*)d_in[9];
  const float* op_b    = (const float*)d_in[10];
  const float* op_w    = (const float*)d_in[11];
  const float* op_bias = (const float*)d_in[12];

  char* ws = (char*)d_ws;
  unsigned short* h     = (unsigned short*)(ws + H_OFF);
  float*          xd    = (float*)(ws + XD_OFF);
  float*          score = (float*)(ws + SC_OFF);
  int*            ti    = (int*)(ws + TI_OFF);
  unsigned short* wt1   = (unsigned short*)(ws + W1_OFF);
  unsigned short* wt2   = (unsigned short*)(ws + W2_OFF);

  const int gemmBlocks = (int)(MROWS / 64);  // 4131

  wconv_kernel<<<256, 256, 0, stream>>>(fp_w, wt1);
  wconv_kernel<<<256, 256, 0, stream>>>(op_w, wt2);

  // feature_processor: LN + GEMM + GELU fused -> h (bf16)
  gemm_fused_kernel<0><<<gemmBlocks, 256, 0, stream>>>(
      x, nullptr, fp_g, fp_b, wt1, fp_bias, nullptr, (void*)h);

  pool_kernel<<<B_ * TR_ * J_, 256, 0, stream>>>(h, xd);
  scorer_kernel<<<B_ * TR_, 256, 0, stream>>>(xd, sc_w1, sc_b1, sc_w2, sc_b2, score);
  topk_kernel<<<B_, 64, 0, stream>>>(score, ti);

  // gather + interpolate + LN + GEMM + GELU + residual fused -> out (fp32)
  gemm_fused_kernel<1><<<gemmBlocks, 256, 0, stream>>>(
      xd, ti, op_g, op_b, wt2, op_bias, x, d_out);
}
